// PeepholeLSTM_60679297958532
// MI455X (gfx1250) — matrix operand order repaired and checked by reading, not compiled, not run on hardware
//
#include <hip/hip_runtime.h>

// Problem sizes (fixed by the reference).
#define BB 32
#define SS 1024
#define DD 512
#define HH 512

typedef __attribute__((ext_vector_type(16))) __bf16        v16bf;
typedef __attribute__((ext_vector_type(8)))  float         v8f;
typedef __attribute__((ext_vector_type(4)))  unsigned int  u32x4;

union Frag {
    u32x4        q[2];   // two 16-byte chunks (global_load_b128 / ds_load_b128)
    unsigned int u[8];
    v16bf        b;      // operand for v_wmma_f32_16x16x32_bf16
};

__device__ __forceinline__ unsigned short f2bf(float f) {
    union { float f; unsigned int u; } x;
    x.f = f;
    unsigned int r = x.u + 0x7FFFu + ((x.u >> 16) & 1u);  // round-to-nearest-even
    return (unsigned short)(r >> 16);
}

__device__ __forceinline__ float sigmoidf_(float x) {
    return 1.0f / (1.0f + __expf(-x));
}

__device__ __forceinline__ v8f vzero8() {
    v8f v;
#pragma unroll
    for (int i = 0; i < 8; ++i) v[i] = 0.0f;
    return v;
}

// ---------------------------------------------------------------------------
// K0: pack the 8 [H=512, K=512] fp32 weight matrices into bf16 WMMA B-fragment
// layout. Packed tile (kt, nt): 32 lanes x 16 bf16; lane l holds N = nt*16 + l%16,
// bf16 s is K = kt*32 + 8*(l/16) + s%8 + 16*(s/8). Also zero both h buffers and
// reset the grid-barrier counter so every launch (incl. replays) is clean.
// ---------------------------------------------------------------------------
__global__ __launch_bounds__(256) void lstm_pack_kernel(
    const float* __restrict__ wxi, const float* __restrict__ wxf,
    const float* __restrict__ wxc, const float* __restrict__ wxo,
    const float* __restrict__ whi, const float* __restrict__ whf,
    const float* __restrict__ whc, const float* __restrict__ who,
    unsigned short* __restrict__ wpx, unsigned short* __restrict__ wph,
    unsigned short* __restrict__ hbuf, int* __restrict__ bar)
{
    unsigned int e = blockIdx.x * 256u + threadIdx.x;   // 0 .. 8*512*512-1
    unsigned int mat  = e >> 18;
    unsigned int r    = e & 262143u;
    unsigned int tile = r >> 9;          // (nt*16 + kt)
    unsigned int rem  = r & 511u;
    unsigned int lane = rem >> 4;
    unsigned int nl   = rem & 15u;
    unsigned int kt   = tile & 15u;
    unsigned int nt   = tile >> 4;
    unsigned int k = kt * 32u + ((lane >> 4) << 3) + (nl & 7u) + ((nl >> 3) << 4);
    unsigned int n = nt * 16u + (lane & 15u);

    const float* W;
    if      (mat == 0u) W = wxi;
    else if (mat == 1u) W = wxf;
    else if (mat == 2u) W = wxc;
    else if (mat == 3u) W = wxo;
    else if (mat == 4u) W = whi;
    else if (mat == 5u) W = whf;
    else if (mat == 6u) W = whc;
    else                W = who;

    unsigned short v = f2bf(W[(size_t)n * 512u + k]);   // W stored [out, in]
    if (mat < 4u) wpx[(size_t)mat * 262144u + r] = v;
    else          wph[(size_t)(mat - 4u) * 262144u + r] = v;

    if (e < 65536u) hbuf[e] = 0;      // zero both h double-buffers (h0 == 0)
    if (e == 0u)    *bar = 0;         // reset grid barrier counter each launch
}

// ---------------------------------------------------------------------------
// K1: input projections. pre[g][s][b][h] = x[b,s,:] @ W_xg^T  (bias in scan).
// Block = 256 thr (8 waves = 2 waves/SIMD) handles one 16-row (b,s)-tile for
// 256 h columns and all 4 gates. x tile staged once in LDS as bf16. Inner
// loop is software pipelined with DOUBLE-BUFFERED fragments; waves_per_eu(2)
// gives the ~512-VGPR budget needed to keep both buffers live.
// ---------------------------------------------------------------------------
__global__ __launch_bounds__(256)
__attribute__((amdgpu_waves_per_eu(2)))
void lstm_xproj_kernel(
    const float* __restrict__ x,              // [B, S, D]
    const unsigned short* __restrict__ wpx,   // [4][512*512] packed bf16
    float* __restrict__ pre)                  // [4][S][B][H]
{
    __shared__ unsigned short xs[16 * 520];   // 512 cols + 8 pad (bank stride)
    const int ROWP = 520;
    const int tid = threadIdx.x;
    const int bs0 = blockIdx.x * 16;          // 0 .. 32767 step 16 (same b)
    const int hgrp = blockIdx.y;              // 0..1

    const float* xrow = x + (size_t)bs0 * DD;
    for (int i = tid; i < 16 * DD; i += 256) {
        int rr = i >> 9, cc = i & 511;
        xs[rr * ROWP + cc] = f2bf(xrow[(size_t)rr * DD + cc]);
    }
    __syncthreads();

    const int wave = tid >> 5, lane = tid & 31;
    const int g = wave & 3;
    const int ntBase = hgrp * 16 + (wave >> 2) * 8;
    const int m  = lane & 15;                 // A row (M)
    const int kh = (lane >> 4) * 8;           // K sub-offset per ISA A layout

    v8f acc[8];
#pragma unroll
    for (int j = 0; j < 8; ++j) acc[j] = vzero8();

    const unsigned short* wg = wpx + (size_t)g * 262144u;

    auto loadA = [&](Frag& a, int kt) {
        const unsigned short* pa = xs + m * ROWP + kt * 32 + kh;
        a.q[0] = *(const u32x4*)(pa);
        a.q[1] = *(const u32x4*)(pa + 16);
    };
    auto loadB8 = [&](Frag* bm, int kt) {
#pragma unroll
        for (int j = 0; j < 8; ++j) {
            const unsigned short* pb =
                wg + ((size_t)((ntBase + j) * 16 + kt) << 9) + (lane << 4);
            bm[j].q[0] = *(const u32x4*)(pb);
            bm[j].q[1] = *(const u32x4*)(pb + 8);
        }
    };
    auto mma8 = [&](Frag& a, Frag* bm) {
#pragma unroll
        for (int j = 0; j < 8; ++j)
            acc[j] = __builtin_amdgcn_wmma_f32_16x16x32_bf16(
                false, a.b, false, bm[j].b, (short)0, acc[j], false, false);
    };

    // double-buffered software pipeline: loads for k+1 issued before mma(k)
    Frag aA, aB;
    Frag bmA[8], bmB[8];
    loadB8(bmA, 0);
    loadA(aA, 0);
#pragma unroll
    for (int kt2 = 0; kt2 < 8; ++kt2) {
        const int k1 = 2 * kt2 + 1;
        loadB8(bmB, k1);
        loadA(aB, k1);
        mma8(aA, bmA);
        if (kt2 < 7) {
            loadB8(bmA, k1 + 1);
            loadA(aA, k1 + 1);
        }
        mma8(aB, bmB);
    }

    // C/D layout: VGPR r, lane -> M = r + 8*(lane>=16), N = lane & 15
    const int b  = bs0 >> 10;
    const int s0 = bs0 & 1023;
    const int nl = lane & 15;
    const int mhi = (lane >> 4) << 3;
    const size_t gbase = (size_t)g * SS * BB * HH;
#pragma unroll
    for (int j = 0; j < 8; ++j) {
        const int h = (ntBase + j) * 16 + nl;
#pragma unroll
        for (int r = 0; r < 8; ++r) {
            const int s = s0 + r + mhi;
            pre[gbase + ((size_t)s * BB + b) * HH + h] = acc[j][r];
        }
    }
}

// ---------------------------------------------------------------------------
// K2: recurrent scan. 8 persistent blocks x 8 waves = 64 waves; wave gw owns
// output tile (mt = gw&1 over B, nt = gw>>1 over H) for ALL 4 gates, so the
// peephole cell state c lives in registers for all 1024 steps.
// Each block copies its 256 KB slice of packed recurrent weights (4 nt-tiles
// x 4 gates) into LDS once (WGP has 320 KB), so the 1024 serial steps read B
// operands via ds_load_b128 instead of hammering L2. h is exchanged through a
// double-buffered bf16 scratch in L2 with one device-scope barrier per step.
// ---------------------------------------------------------------------------
__global__ __launch_bounds__(256)
__attribute__((amdgpu_waves_per_eu(2)))
void lstm_scan_kernel(
    const float* __restrict__ pre,            // [4][S][B][H]
    const unsigned short* __restrict__ wph,   // [4][512*512] packed bf16
    const float* __restrict__ Wci, const float* __restrict__ Wcf,
    const float* __restrict__ Wco,
    const float* __restrict__ bi,  const float* __restrict__ bf,
    const float* __restrict__ bc,  const float* __restrict__ bo,
    unsigned short* __restrict__ hbuf,        // [2][B*H] bf16 double buffer
    int* __restrict__ bar,
    float* __restrict__ out)                  // [B*S*H] ++ h_T[B*H] ++ c_T[B*H]
{
    extern __shared__ unsigned short wlds[];  // 131072 ushorts = 256 KB
    const int tid  = threadIdx.x;
    const int wave = tid >> 5, lane = tid & 31;
    const int blk  = blockIdx.x;
    const int gw = blk * 8 + wave;            // 0..63
    const int mt = gw & 1;                    // B tile (rows 0-15 / 16-31)
    const int nt = gw >> 1;                   // H tile 0..31
    const int ntl = wave >> 1;                // local H tile within block: 0..3
    const int nl  = lane & 15;
    const int mhi = (lane >> 4) << 3;
    const int h = nt * 16 + nl;
    const int arow = mt * 16 + (lane & 15);   // A fragment row in h matrix
    const int kh   = (lane >> 4) * 8;

    // Stage this block's weight slice into LDS.
    // LDS tile layout: base = ((g*4 + ntl)*16 + kt)*512; chunk (lane, half) at
    // base + half*256 + lane*8 (ushorts) -> 16B-stride ds_load_b128 (2-way bc).
    for (int ci = tid; ci < 16384; ci += 256) {
        const int half = ci & 1;
        const int ln   = (ci >> 1) & 31;
        const int kt   = (ci >> 6) & 15;
        const int ntc  = (ci >> 10) & 3;
        const int g    = (ci >> 12) & 3;
        const unsigned short* src =
            wph + (size_t)g * 262144u +
            ((size_t)((blk * 4 + ntc) * 16 + kt) << 9) + (ln << 4) + (half << 3);
        unsigned short* dst =
            wlds + (((g * 4 + ntc) * 16 + kt) << 9) + (half << 8) + (ln << 3);
        *(u32x4*)dst = *(const u32x4*)src;
    }
    __syncthreads();

    // per-lane gate constants (one h per lane for this tile)
    const float wci = Wci[h], wcf = Wcf[h], wco = Wco[h];
    const float vbi = bi[h], vbf = bf[h], vbc = bc[h], vbo = bo[h];

    float c[8];
#pragma unroll
    for (int r = 0; r < 8; ++r) c[r] = 0.0f;

    const size_t SBH = (size_t)SS * BB * HH;
    const size_t OUT_T = (size_t)BB * SS * HH;

    v8f ai, af, ac, ao;

    auto loadB4 = [&](Frag* bm, int kt) {
#pragma unroll
        for (int g = 0; g < 4; ++g) {
            const unsigned short* pb = wlds + (((g * 4 + ntl) * 16 + kt) << 9);
            bm[g].q[0] = *(const u32x4*)(pb + (lane << 3));
            bm[g].q[1] = *(const u32x4*)(pb + 256 + (lane << 3));
        }
    };
    auto mma4 = [&](Frag& a, Frag* bm) {
        ai = __builtin_amdgcn_wmma_f32_16x16x32_bf16(false, a.b, false, bm[0].b, (short)0, ai, false, false);
        af = __builtin_amdgcn_wmma_f32_16x16x32_bf16(false, a.b, false, bm[1].b, (short)0, af, false, false);
        ac = __builtin_amdgcn_wmma_f32_16x16x32_bf16(false, a.b, false, bm[2].b, (short)0, ac, false, false);
        ao = __builtin_amdgcn_wmma_f32_16x16x32_bf16(false, a.b, false, bm[3].b, (short)0, ao, false, false);
    };

    for (int t = 0; t < SS; ++t) {
        const unsigned short* hsrc = hbuf + (size_t)(t & 1) * (BB * HH);
        unsigned short*       hdst = hbuf + (size_t)((t & 1) ^ 1) * (BB * HH);

        auto loadA = [&](Frag& a, int kt) {
            const unsigned short* pa = hsrc + arow * 512 + kt * 32 + kh;
            a.q[0] = *(const u32x4*)(pa);
            a.q[1] = *(const u32x4*)(pa + 16);
        };

        // hoist this step's input-projection loads ahead of the WMMA chain
        float xi[8], xfv[8], xcv[8], xov[8];
        {
            const float* p0 = pre + ((size_t)t * BB) * HH + h;
#pragma unroll
            for (int r = 0; r < 8; ++r) {
                const size_t o = (size_t)(mt * 16 + mhi + r) * HH;
                xi[r]  = p0[0 * SBH + o];
                xfv[r] = p0[1 * SBH + o];
                xcv[r] = p0[2 * SBH + o];
                xov[r] = p0[3 * SBH + o];
            }
        }
        // prefetch next step's projections into cache (global_prefetch_b8)
        if (t + 1 < SS) {
            const float* pn = pre + ((size_t)(t + 1) * BB) * HH + h;
#pragma unroll
            for (int r = 0; r < 8; ++r) {
                const size_t o = (size_t)(mt * 16 + mhi + r) * HH;
                __builtin_prefetch(pn + 0 * SBH + o, 0, 1);
                __builtin_prefetch(pn + 1 * SBH + o, 0, 1);
                __builtin_prefetch(pn + 2 * SBH + o, 0, 1);
                __builtin_prefetch(pn + 3 * SBH + o, 0, 1);
            }
        }

        ai = vzero8(); af = vzero8(); ac = vzero8(); ao = vzero8();

        // double-buffered pipeline: fragment loads for k+1 precede mma(k)
        Frag aA, aB;
        Frag bmA[4], bmB[4];
        loadB4(bmA, 0);
        loadA(aA, 0);
#pragma unroll
        for (int kt2 = 0; kt2 < 8; ++kt2) {
            const int k1 = 2 * kt2 + 1;
            loadB4(bmB, k1);
            loadA(aB, k1);
            mma4(aA, bmA);
            if (kt2 < 7) {
                loadB4(bmA, k1 + 1);
                loadA(aA, k1 + 1);
            }
            mma4(aB, bmB);
        }

        // peephole LSTM cell update (c stays in registers)
#pragma unroll
        for (int r = 0; r < 8; ++r) {
            const int bb = mt * 16 + mhi + r;
            const float ig = sigmoidf_(xi[r]  + ai[r] + wci * c[r] + vbi);
            const float fg = sigmoidf_(xfv[r] + af[r] + wcf * c[r] + vbf);
            const float ch = tanhf(xcv[r] + ac[r] + vbc);
            const float cn = fg * c[r] + ig * ch;
            const float og = sigmoidf_(xov[r] + ao[r] + wco * cn + vbo);
            const float hn = og * tanhf(cn);
            c[r] = cn;
            out[((size_t)bb * SS + t) * HH + h] = hn;     // hs -> [B,S,H]
            hdst[bb * 512 + h] = f2bf(hn);
            if (t == SS - 1) {
                out[OUT_T + (size_t)bb * HH + h] = hn;                      // h_T
                out[OUT_T + (size_t)BB * HH + (size_t)bb * HH + h] = cn;    // c_T
            }
        }

        // device-scope step barrier (monotonic counter, 8 blocks)
        __builtin_amdgcn_fence(__ATOMIC_RELEASE, "agent");
        __syncthreads();
        if (tid == 0) {
            __hip_atomic_fetch_add(bar, 1, __ATOMIC_RELEASE, __HIP_MEMORY_SCOPE_AGENT);
            const int target = 8 * (t + 1);
            while (__hip_atomic_load(bar, __ATOMIC_ACQUIRE, __HIP_MEMORY_SCOPE_AGENT) < target) {
                __builtin_amdgcn_s_sleep(1);
            }
        }
        __syncthreads();
        __builtin_amdgcn_fence(__ATOMIC_ACQUIRE, "agent");
    }
}

// ---------------------------------------------------------------------------
extern "C" void kernel_launch(void* const* d_in, const int* in_sizes, int n_in,
                              void* d_out, int out_size, void* d_ws, size_t ws_size,
                              hipStream_t stream) {
    (void)in_sizes; (void)n_in; (void)out_size; (void)ws_size;

    const float* x    = (const float*)d_in[0];
    const float* W_xi = (const float*)d_in[1];
    const float* W_hi = (const float*)d_in[2];
    const float* W_ci = (const float*)d_in[3];
    const float* W_xf = (const float*)d_in[4];
    const float* W_hf = (const float*)d_in[5];
    const float* W_cf = (const float*)d_in[6];
    const float* W_xc = (const float*)d_in[7];
    const float* W_hc = (const float*)d_in[8];
    const float* W_xo = (const float*)d_in[9];
    const float* W_ho = (const float*)d_in[10];
    const float* W_co = (const float*)d_in[11];
    const float* b_i  = (const float*)d_in[12];
    const float* b_f  = (const float*)d_in[13];
    const float* b_c  = (const float*)d_in[14];
    const float* b_o  = (const float*)d_in[15];

    // workspace layout
    char* ws = (char*)d_ws;
    unsigned short* wpx  = (unsigned short*)(ws);                 // 2 MB
    unsigned short* wph  = (unsigned short*)(ws + 2097152);       // 2 MB
    unsigned short* hbuf = (unsigned short*)(ws + 4194304);       // 128 KB
    int*            bar  = (int*)(ws + 4325376);                  // 512 B slot
    float*          pre  = (float*)(ws + 4325888);                // 256 MB

    // allow 256 KB dynamic LDS for the scan kernel (WGP has 320 KB)
    (void)hipFuncSetAttribute((const void*)lstm_scan_kernel,
                              hipFuncAttributeMaxDynamicSharedMemorySize,
                              262144);

    // K0: pack weights to bf16 WMMA fragments + reset h buffers / barrier
    lstm_pack_kernel<<<8192, 256, 0, stream>>>(
        W_xi, W_xf, W_xc, W_xo, W_hi, W_hf, W_hc, W_ho, wpx, wph, hbuf, bar);

    // K1: input projections (bf16 WMMA, fp32 accumulate)
    lstm_xproj_kernel<<<dim3(2048, 2), 256, 0, stream>>>(x, wpx, pre);

    // K2: persistent recurrent scan (256 KB LDS-resident weights per block)
    lstm_scan_kernel<<<8, 256, 262144, stream>>>(
        pre, wph, W_ci, W_cf, W_co, b_i, b_f, b_c, b_o,
        hbuf, bar, (float*)d_out);
}
